// MultiHeadAttention_9534827397446
// MI455X (gfx1250) — matrix operand order repaired and checked by reading, not compiled, not run on hardware
//
#include <hip/hip_runtime.h>
#include <hip/hip_bf16.h>
#include <stdint.h>

#define D_MODEL  768
#define N_HEADS  4
#define HEAD_DIM 192
#define BATCH    4
#define SEQ      2048
#define BS       (BATCH * SEQ)

typedef __attribute__((ext_vector_type(16))) __bf16 v16bf;
typedef __attribute__((ext_vector_type(8)))  float  v8f;

// ---------- helpers ----------

__device__ __forceinline__ unsigned short f2bf(float f) {
  union { float f; unsigned u; } a; a.f = f;
  unsigned u = a.u;
  u += 0x7FFFu + ((u >> 16) & 1u);   // round-to-nearest-even
  return (unsigned short)(u >> 16);
}

union ABFrag { v16bf v; unsigned u[8]; };

// A-matrix fragment (16x32 bf16), source row-major [16][lda], base at (m0,k0).
// ISA 7.12.2: lanes 0-15: M=lane, K=0..7 (v0-3), 16..23 (v4-7);
// lanes 16-31: M=lane-16, K=8..15 (v0-3), 24..31 (v4-7). K-pairs packed per VGPR.
__device__ __forceinline__ v16bf load_a_frag(const unsigned short* base, int lda) {
  int lane = threadIdx.x & 31;
  int m    = lane & 15;
  int koff = (lane >> 4) << 3;  // 0 or 8
  ABFrag r;
  const unsigned short* row = base + m * lda + koff;
#pragma unroll
  for (int v = 0; v < 4; ++v) {
    r.u[v]     = *(const unsigned*)(row + 2 * v);
    r.u[4 + v] = *(const unsigned*)(row + 16 + 2 * v);
  }
  return r.v;
}

// B-matrix fragment (32x16 bf16), source is B^T row-major [16][ldb] (column-major B),
// base at (n0,k0). lanes 0-15: N=lane, K=0..7 (v0-3), 16..23 (v4-7);
// lanes 16-31: N=lane-16, K=8..15 (v0-3), 24..31 (v4-7).
__device__ __forceinline__ v16bf load_b_frag(const unsigned short* base, int ldb) {
  int lane = threadIdx.x & 31;
  int n    = lane & 15;
  int koff = (lane >> 4) << 3;  // 0 or 8
  ABFrag r;
  const unsigned short* row = base + n * ldb + koff;
#pragma unroll
  for (int v = 0; v < 4; ++v) {
    r.u[v]     = *(const unsigned*)(row + 2 * v);
    r.u[4 + v] = *(const unsigned*)(row + 16 + 2 * v);
  }
  return r.v;
}

__device__ __forceinline__ v8f wmma_bf16(v16bf a, v16bf b, v8f c) {
  return __builtin_amdgcn_wmma_f32_16x16x32_bf16(false, a, false, b, (short)0, c,
                                                 false, false);
}

// ---------- pack kernels ----------

__global__ void pack_x_kernel(const float* __restrict__ x,
                              unsigned short* __restrict__ xb, int n) {
  int i = blockIdx.x * blockDim.x + threadIdx.x;
  if (i < n) xb[i] = f2bf(x[i]);
}

// Transpose + convert weights: W [in][out] f32 -> W^T [out][in] bf16
__global__ void pack_wT_kernel(const float* W0, const float* W1,
                               const float* W2, const float* W3,
                               unsigned short* T0, unsigned short* T1,
                               unsigned short* T2, unsigned short* T3) {
  const float* W = (blockIdx.y == 0) ? W0 : (blockIdx.y == 1) ? W1
                 : (blockIdx.y == 2) ? W2 : W3;
  unsigned short* T = (blockIdx.y == 0) ? T0 : (blockIdx.y == 1) ? T1
                    : (blockIdx.y == 2) ? T2 : T3;
  int i = blockIdx.x * blockDim.x + threadIdx.x;
  if (i < D_MODEL * D_MODEL) {
    int r = i / D_MODEL;  // in
    int c = i % D_MODEL;  // out
    T[c * D_MODEL + r] = f2bf(W[i]);
  }
}

// ---------- fused QKV projection ----------
// Each wave: 16 rows x 64 cols; double-buffered k-loop.
// Grid: (768 blocks x 8 waves) = 512 rowblks * 12 colblks; blockIdx.y selects Q/K/V.
// Q,K -> [b,h,s,d]; V -> transposed [b,h,d,s].
__global__ void qkv_proj_kernel(const unsigned short* __restrict__ xb,
                                const unsigned short* __restrict__ WqT,
                                const unsigned short* __restrict__ WkT,
                                const unsigned short* __restrict__ WvT,
                                const float* __restrict__ bq,
                                const float* __restrict__ bk,
                                const float* __restrict__ bv,
                                unsigned short* __restrict__ Qb,
                                unsigned short* __restrict__ Kb,
                                unsigned short* __restrict__ Vt) {
  int which = blockIdx.y;
  const unsigned short* Wt = (which == 0) ? WqT : (which == 1) ? WkT : WvT;
  const float* bias        = (which == 0) ? bq  : (which == 1) ? bk  : bv;
  unsigned short* out      = (which == 0) ? Qb  : (which == 1) ? Kb  : Vt;

  int gw = blockIdx.x * (blockDim.x >> 5) + (threadIdx.x >> 5);
  int rowblk = gw / 12, colblk = gw % 12;
  int m0 = rowblk * 16, n0 = colblk * 64;

  v8f acc[4];
#pragma unroll
  for (int j = 0; j < 4; ++j)
#pragma unroll
    for (int v = 0; v < 8; ++v) acc[j][v] = 0.0f;

  const unsigned short* Abase = xb + (size_t)m0 * D_MODEL;

  // software pipeline: fetch k0+32 fragments while WMMAs consume k0
  v16bf a0 = load_a_frag(Abase, D_MODEL);
  v16bf b0[4];
#pragma unroll
  for (int j = 0; j < 4; ++j)
    b0[j] = load_b_frag(Wt + (size_t)(n0 + 16 * j) * D_MODEL, D_MODEL);

  for (int k0 = 0; k0 < D_MODEL; k0 += 32) {
    int kn = k0 + 32;
    v16bf a1;
    v16bf b1[4];
    if (kn < D_MODEL) {
      a1 = load_a_frag(Abase + kn, D_MODEL);
#pragma unroll
      for (int j = 0; j < 4; ++j)
        b1[j] = load_b_frag(Wt + (size_t)(n0 + 16 * j) * D_MODEL + kn, D_MODEL);
    }
#pragma unroll
    for (int j = 0; j < 4; ++j) acc[j] = wmma_bf16(a0, b0[j], acc[j]);
    if (kn < D_MODEL) {
      a0 = a1;
#pragma unroll
      for (int j = 0; j < 4; ++j) b0[j] = b1[j];
    }
  }

  int lane = threadIdx.x & 31;
  int nloc = lane & 15;
  int mb   = (lane >> 4) * 8;
#pragma unroll
  for (int j = 0; j < 4; ++j) {
#pragma unroll
    for (int v = 0; v < 8; ++v) {
      int m = m0 + mb + v;
      int n = n0 + 16 * j + nloc;
      float val = acc[j][v] + bias[n];
      int b_ = m / SEQ, s = m % SEQ;
      int h = n / HEAD_DIM, dd = n % HEAD_DIM;
      size_t idx = (which == 2)
          ? ((size_t)(b_ * N_HEADS + h) * HEAD_DIM + dd) * SEQ + s
          : ((size_t)(b_ * N_HEADS + h) * SEQ + s) * HEAD_DIM + dd;
      out[idx] = f2bf(val);
    }
  }
}

// ---------- causal flash attention ----------
// One wave owns 16 queries x Dh=192, streams keys in steps of 32 with online softmax.
__global__ void attn_kernel(const unsigned short* __restrict__ Qb,
                            const unsigned short* __restrict__ Kb,
                            const unsigned short* __restrict__ Vt,
                            unsigned short* __restrict__ ctx) {
  __shared__ unsigned short pbuf[8][16 * 32];  // per-wave P scratch (bf16)

  int wave = threadIdx.x >> 5;
  int lane = threadIdx.x & 31;
  int gw   = blockIdx.x * (blockDim.x >> 5) + wave;
  int qblk = gw & 127;       // S/16 = 128 query blocks
  int bh   = gw >> 7;        // 0..15
  int b_   = bh >> 2, h = bh & 3;

  const unsigned short* Qp = Qb + (size_t)bh * SEQ * HEAD_DIM;
  const unsigned short* Kp = Kb + (size_t)bh * SEQ * HEAD_DIM;
  const unsigned short* Vp = Vt + (size_t)bh * HEAD_DIM * SEQ;  // [Dh][S]
  int q0 = qblk * 16;

  // Q fragments resident in registers for all key steps
  v16bf qf[6];
#pragma unroll
  for (int c = 0; c < 6; ++c)
    qf[c] = load_a_frag(Qp + (size_t)q0 * HEAD_DIM + 32 * c, HEAD_DIM);

  v8f o[12];
#pragma unroll
  for (int j = 0; j < 12; ++j)
#pragma unroll
    for (int v = 0; v < 8; ++v) o[j][v] = 0.0f;

  float mrow[8], lrow[8];
#pragma unroll
  for (int v = 0; v < 8; ++v) { mrow[v] = -1e30f; lrow[v] = 0.0f; }

  int nloc = lane & 15;
  int mb   = (lane >> 4) * 8;
  const float scale = 0.0721687836487032f;  // 1/sqrt(192)
  unsigned short* P = pbuf[wave];

  for (int k0 = 0; k0 <= q0; k0 += 32) {
    // prefetch next key step: 32 K rows (one per lane), 6x32 V^T rows at next cols
    if (k0 + 32 <= q0) {
      __builtin_prefetch(Kp + (size_t)(k0 + 32 + lane) * HEAD_DIM, 0, 0);
#pragma unroll
      for (int t = 0; t < 6; ++t)
        __builtin_prefetch(Vp + (size_t)(t * 32 + lane) * SEQ + (k0 + 32), 0, 0);
    }

    // scores S[16q x 32k] = Q . K^T  (two 16x16 C fragments), K frags double-buffered
    v8f sf0, sf1;
#pragma unroll
    for (int v = 0; v < 8; ++v) { sf0[v] = 0.0f; sf1[v] = 0.0f; }

    v16bf kf0 = load_b_frag(Kp + (size_t)k0 * HEAD_DIM, HEAD_DIM);
    v16bf kf1 = load_b_frag(Kp + (size_t)(k0 + 16) * HEAD_DIM, HEAD_DIM);
#pragma unroll
    for (int c = 0; c < 6; ++c) {
      v16bf nf0, nf1;
      if (c < 5) {
        nf0 = load_b_frag(Kp + (size_t)k0 * HEAD_DIM + 32 * (c + 1), HEAD_DIM);
        nf1 = load_b_frag(Kp + (size_t)(k0 + 16) * HEAD_DIM + 32 * (c + 1), HEAD_DIM);
      }
      sf0 = wmma_bf16(qf[c], kf0, sf0);
      sf1 = wmma_bf16(qf[c], kf1, sf1);
      if (c < 5) { kf0 = nf0; kf1 = nf1; }
    }

    // online softmax (rows spread over 16 lanes of each half)
    float p0[8], p1[8], corr[8];
#pragma unroll
    for (int v = 0; v < 8; ++v) {
      int q = q0 + mb + v;
      float s0 = sf0[v] * scale; if (k0 + nloc > q)      s0 = -1e30f;
      float s1 = sf1[v] * scale; if (k0 + 16 + nloc > q) s1 = -1e30f;
      float rm = fmaxf(s0, s1);
      rm = fmaxf(rm, __shfl_xor(rm, 1));
      rm = fmaxf(rm, __shfl_xor(rm, 2));
      rm = fmaxf(rm, __shfl_xor(rm, 4));
      rm = fmaxf(rm, __shfl_xor(rm, 8));
      float mnew = fmaxf(mrow[v], rm);
      corr[v] = __expf(mrow[v] - mnew);
      float e0 = __expf(s0 - mnew);
      float e1 = __expf(s1 - mnew);
      float rs = e0 + e1;
      rs += __shfl_xor(rs, 1);
      rs += __shfl_xor(rs, 2);
      rs += __shfl_xor(rs, 4);
      rs += __shfl_xor(rs, 8);
      lrow[v] = lrow[v] * corr[v] + rs;
      mrow[v] = mnew;
      p0[v] = e0; p1[v] = e1;
    }
#pragma unroll
    for (int j = 0; j < 12; ++j)
#pragma unroll
      for (int v = 0; v < 8; ++v) o[j][v] *= corr[v];

    // C-layout P -> row-major bf16 in LDS, then reload as A fragment
#pragma unroll
    for (int v = 0; v < 8; ++v) {
      int m = mb + v;
      P[m * 32 + nloc]      = f2bf(p0[v]);
      P[m * 32 + 16 + nloc] = f2bf(p1[v]);
    }
    asm volatile("s_wait_dscnt 0x0" ::: "memory");  // same-wave LDS RAW

    ABFrag r;
    {
      int m    = lane & 15;
      int koff = (lane >> 4) << 3;
      const unsigned short* row = P + m * 32 + koff;
#pragma unroll
      for (int v = 0; v < 4; ++v) {
        r.u[v]     = *(const unsigned*)(row + 2 * v);
        r.u[4 + v] = *(const unsigned*)(row + 16 + 2 * v);
      }
    }
    v16bf pf = r.v;

    // O += P . V  (V^T is row-major [Dh][S] -> direct B fragments), double-buffered
    v16bf vf = load_b_frag(Vp + k0, SEQ);
#pragma unroll
    for (int j = 0; j < 12; ++j) {
      v16bf vn;
      if (j < 11)
        vn = load_b_frag(Vp + (size_t)(16 * (j + 1)) * SEQ + k0, SEQ);
      o[j] = wmma_bf16(pf, vf, o[j]);
      if (j < 11) vf = vn;
    }
  }

  // normalize and emit ctx[b][s][h*Dh+d] in bf16
#pragma unroll
  for (int v = 0; v < 8; ++v) lrow[v] = 1.0f / lrow[v];
#pragma unroll
  for (int j = 0; j < 12; ++j) {
#pragma unroll
    for (int v = 0; v < 8; ++v) {
      int sq = q0 + mb + v;
      int dd = 16 * j + nloc;
      ctx[((size_t)b_ * SEQ + sq) * D_MODEL + h * HEAD_DIM + dd] =
          f2bf(o[j][v] * lrow[v]);
    }
  }
}

// ---------- output projection ----------
__global__ void out_proj_kernel(const unsigned short* __restrict__ ctx,
                                const unsigned short* __restrict__ WoT,
                                const float* __restrict__ bo,
                                float* __restrict__ out) {
  int gw = blockIdx.x * (blockDim.x >> 5) + (threadIdx.x >> 5);
  int rowblk = gw / 12, colblk = gw % 12;
  int m0 = rowblk * 16, n0 = colblk * 64;

  v8f acc[4];
#pragma unroll
  for (int j = 0; j < 4; ++j)
#pragma unroll
    for (int v = 0; v < 8; ++v) acc[j][v] = 0.0f;

  const unsigned short* Abase = ctx + (size_t)m0 * D_MODEL;

  v16bf a0 = load_a_frag(Abase, D_MODEL);
  v16bf b0[4];
#pragma unroll
  for (int j = 0; j < 4; ++j)
    b0[j] = load_b_frag(WoT + (size_t)(n0 + 16 * j) * D_MODEL, D_MODEL);

  for (int k0 = 0; k0 < D_MODEL; k0 += 32) {
    int kn = k0 + 32;
    v16bf a1;
    v16bf b1[4];
    if (kn < D_MODEL) {
      a1 = load_a_frag(Abase + kn, D_MODEL);
#pragma unroll
      for (int j = 0; j < 4; ++j)
        b1[j] = load_b_frag(WoT + (size_t)(n0 + 16 * j) * D_MODEL + kn, D_MODEL);
    }
#pragma unroll
    for (int j = 0; j < 4; ++j) acc[j] = wmma_bf16(a0, b0[j], acc[j]);
    if (kn < D_MODEL) {
      a0 = a1;
#pragma unroll
      for (int j = 0; j < 4; ++j) b0[j] = b1[j];
    }
  }

  int lane = threadIdx.x & 31;
  int nloc = lane & 15;
  int mb   = (lane >> 4) * 8;
#pragma unroll
  for (int j = 0; j < 4; ++j) {
#pragma unroll
    for (int v = 0; v < 8; ++v) {
      int m = m0 + mb + v;
      int n = n0 + 16 * j + nloc;
      out[(size_t)m * D_MODEL + n] = acc[j][v] + bo[n];
    }
  }
}

// ---------- launch ----------

extern "C" void kernel_launch(void* const* d_in, const int* in_sizes, int n_in,
                              void* d_out, int out_size, void* d_ws, size_t ws_size,
                              hipStream_t stream) {
  (void)in_sizes; (void)n_in; (void)out_size; (void)ws_size;

  const float* x  = (const float*)d_in[0];
  const float* Wq = (const float*)d_in[1];
  const float* bq = (const float*)d_in[2];
  const float* Wk = (const float*)d_in[3];
  const float* bk = (const float*)d_in[4];
  const float* Wv = (const float*)d_in[5];
  const float* bv = (const float*)d_in[6];
  const float* Wo = (const float*)d_in[7];
  const float* bo = (const float*)d_in[8];
  float* out = (float*)d_out;

  char* ws = (char*)d_ws;
  size_t off = 0;
  auto carve = [&](size_t bytes) {
    char* p = ws + off;
    off += (bytes + 255) & ~(size_t)255;
    return p;
  };
  const size_t XB = (size_t)BS * D_MODEL * 2;       // bf16 activations
  const size_t WT = (size_t)D_MODEL * D_MODEL * 2;  // bf16 weight
  const size_t QS = (size_t)BATCH * N_HEADS * SEQ * HEAD_DIM * 2;

  unsigned short* xb   = (unsigned short*)carve(XB);
  unsigned short* WqT  = (unsigned short*)carve(WT);
  unsigned short* WkT  = (unsigned short*)carve(WT);
  unsigned short* WvT  = (unsigned short*)carve(WT);
  unsigned short* WoT  = (unsigned short*)carve(WT);
  unsigned short* Qb   = (unsigned short*)carve(QS);
  unsigned short* Kb   = (unsigned short*)carve(QS);
  unsigned short* Vt   = (unsigned short*)carve(QS);
  unsigned short* ctxb = (unsigned short*)carve(XB);

  int n = BS * D_MODEL;
  pack_x_kernel<<<(n + 255) / 256, 256, 0, stream>>>(x, xb, n);

  dim3 gpw((D_MODEL * D_MODEL + 255) / 256, 4);
  pack_wT_kernel<<<gpw, 256, 0, stream>>>(Wq, Wk, Wv, Wo, WqT, WkT, WvT, WoT);

  // 512 rowblks * 12 colblks = 6144 waves = 768 blocks of 8 waves; y = Q/K/V
  qkv_proj_kernel<<<dim3(768, 3), 256, 0, stream>>>(xb, WqT, WkT, WvT,
                                                    bq, bk, bv, Qb, Kb, Vt);

  // 16 (b,h) * 128 qblks = 2048 waves = 256 blocks of 8 waves
  attn_kernel<<<256, 256, 0, stream>>>(Qb, Kb, Vt, ctxb);

  out_proj_kernel<<<768, 256, 0, stream>>>(ctxb, WoT, bo, out);
}
